// LowRankMahalanobisDistance_3058016715342
// MI455X (gfx1250) — matrix operand order repaired and checked by reading, not compiled, not run on hardware
//
#include <hip/hip_runtime.h>

typedef __attribute__((ext_vector_type(4)))  float  v4f;
typedef __attribute__((ext_vector_type(8)))  float  v8f;
typedef __attribute__((ext_vector_type(8)))  __bf16 v8bf;
typedef __attribute__((ext_vector_type(16))) __bf16 v16bf;

#define NROWS 8192
#define DIM   1024
#define RANK  128

static __device__ inline v8f wmma_bf16(v16bf a, v16bf b, v8f c) {
  // D = A(16x32 bf16) * B(32x16 bf16) + C(f32)
  return __builtin_amdgcn_wmma_f32_16x16x32_bf16(false, a, false, b, (short)0, c,
                                                 false, false);
}

// A-fragment (16-bit A 16x32): lane holds row M=lane%16; K chunks
// [8*half, 8*half+8) and [16+8*half, 16+8*half+8)   (half = lane>>4)
static __device__ inline v16bf load_fragA(const __bf16* rowp, int k, int half) {
  v8bf c0 = *(const v8bf*)(rowp + k + 8 * half);
  v8bf c1 = *(const v8bf*)(rowp + k + 16 + 8 * half);
  return __builtin_shufflevector(c0, c1, 0, 1, 2, 3, 4, 5, 6, 7,
                                 8, 9, 10, 11, 12, 13, 14, 15);
}

// B-fragment (16-bit B 32x16): lane holds col N=lane%16; K chunks
// [8*half, 8*half+8) and [16+8*half, 16+8*half+8)
static __device__ inline v16bf load_fragB(const __bf16* rowp, int k, int half) {
  v8bf c0 = *(const v8bf*)(rowp + k + 8 * half);
  v8bf c1 = *(const v8bf*)(rowp + k + 16 + 8 * half);
  return __builtin_shufflevector(c0, c1, 0, 1, 2, 3, 4, 5, 6, 7,
                                 8, 9, 10, 11, 12, 13, 14, 15);
}

// ---------------------------------------------------------------------------
// Kernel 1: split-bf16 projection  xL = x @ L^T  (and yL = y @ L^T).
// Workgroup = 8 waves; wave w computes rows [blk*128 + w*16, +16) x all 128
// rank columns. f32 inputs are loaded as b128 and split into bf16 hi/lo in
// registers to feed 3 bf16 WMMAs per 16x16 tile per K-step (bf16x3 ~ f32).
// ---------------------------------------------------------------------------
__global__ __launch_bounds__(256) void proj_split_kernel(
    const float* __restrict__ x, const float* __restrict__ y,
    const float* __restrict__ L,
    __bf16* __restrict__ xh, __bf16* __restrict__ xl,
    __bf16* __restrict__ yh, __bf16* __restrict__ yl) {
  const float* src = blockIdx.y ? y : x;
  __bf16* oh = blockIdx.y ? yh : xh;
  __bf16* ol = blockIdx.y ? yl : xl;

  const int lane = threadIdx.x & 31;
  const int wave = threadIdx.x >> 5;
  const int half = lane >> 4;
  const int li   = lane & 15;
  const int mWave = blockIdx.x * 128 + wave * 16;

  v8f acc[8];
#pragma unroll
  for (int t = 0; t < 8; ++t) acc[t] = (v8f){0.f, 0.f, 0.f, 0.f, 0.f, 0.f, 0.f, 0.f};

  const float* aRow = src + (size_t)(mWave + li) * DIM;

  for (int k = 0; k < DIM; k += 32) {
    // --- A fragment from f32 x/y rows, split into bf16 hi/lo ---
    float av[16];
    {
      v4f t0 = *(const v4f*)(aRow + k + 8 * half);
      v4f t1 = *(const v4f*)(aRow + k + 8 * half + 4);
      v4f t2 = *(const v4f*)(aRow + k + 16 + 8 * half);
      v4f t3 = *(const v4f*)(aRow + k + 16 + 8 * half + 4);
#pragma unroll
      for (int i = 0; i < 4; ++i) {
        av[i] = t0[i]; av[4 + i] = t1[i]; av[8 + i] = t2[i]; av[12 + i] = t3[i];
      }
    }
    v16bf ahF, alF;
#pragma unroll
    for (int i = 0; i < 16; ++i) {
      float v = av[i];
      __bf16 h = (__bf16)v;
      ahF[i] = h;
      alF[i] = (__bf16)(v - (float)h);
    }

    // --- B fragments from L (rank index = column), 8 N-tiles ---
#pragma unroll
    for (int nt = 0; nt < 8; ++nt) {
      const float* bRow = L + (size_t)(nt * 16 + li) * DIM + k + 8 * half;
      float bv[16];
      v4f u0 = *(const v4f*)(bRow);
      v4f u1 = *(const v4f*)(bRow + 4);
      v4f u2 = *(const v4f*)(bRow + 16);
      v4f u3 = *(const v4f*)(bRow + 20);
#pragma unroll
      for (int i = 0; i < 4; ++i) {
        bv[i] = u0[i]; bv[4 + i] = u1[i]; bv[8 + i] = u2[i]; bv[12 + i] = u3[i];
      }
      v16bf bhF, blF;
#pragma unroll
      for (int i = 0; i < 16; ++i) {
        float v = bv[i];
        __bf16 h = (__bf16)v;
        bhF[i] = h;
        blF[i] = (__bf16)(v - (float)h);
      }
      acc[nt] = wmma_bf16(ahF, bhF, acc[nt]);
      acc[nt] = wmma_bf16(ahF, blF, acc[nt]);
      acc[nt] = wmma_bf16(alF, bhF, acc[nt]);
    }
  }

  // Epilogue: split f32 projection into bf16 hi/lo for the main GEMM.
#pragma unroll
  for (int nt = 0; nt < 8; ++nt) {
    const int n = nt * 16 + li;
#pragma unroll
    for (int i = 0; i < 8; ++i) {
      const int m = mWave + 8 * half + i;
      float v = acc[nt][i];
      __bf16 h = (__bf16)v;
      oh[(size_t)m * RANK + n] = h;
      ol[(size_t)m * RANK + n] = (__bf16)(v - (float)h);
    }
  }
}

// ---------------------------------------------------------------------------
// Kernel 2: per-row squared norms of the projections (hi+lo reconstructs f32).
// ---------------------------------------------------------------------------
__global__ __launch_bounds__(256) void rownorm_kernel(
    const __bf16* __restrict__ xh, const __bf16* __restrict__ xl,
    const __bf16* __restrict__ yh, const __bf16* __restrict__ yl,
    float* __restrict__ xn, float* __restrict__ yn) {
  const __bf16* h = blockIdx.y ? yh : xh;
  const __bf16* l = blockIdx.y ? yl : xl;
  float* o = blockIdx.y ? yn : xn;

  const int r = blockIdx.x * blockDim.x + threadIdx.x;
  if (r >= NROWS) return;
  const __bf16* ph = h + (size_t)r * RANK;
  const __bf16* pl = l + (size_t)r * RANK;
  float s = 0.f;
#pragma unroll
  for (int k = 0; k < RANK; k += 8) {
    v8bf a = *(const v8bf*)(ph + k);
    v8bf b = *(const v8bf*)(pl + k);
#pragma unroll
    for (int i = 0; i < 8; ++i) {
      float v = (float)a[i] + (float)b[i];
      s += v * v;
    }
  }
  o[r] = s;
}

// ---------------------------------------------------------------------------
// Kernel 3: cross = xL @ yL^T via split-bf16 WMMA; fused distance epilogue.
// Workgroup tile 128(M) x 64(N); 8 waves as 4x2; wave tile 32x32 (4 accs).
// K = 128 -> 4 steps of 32; 3 WMMAs per tile pair per step (hh + hl + lh).
// ---------------------------------------------------------------------------
__global__ __launch_bounds__(256) void cross_kernel(
    const __bf16* __restrict__ xh, const __bf16* __restrict__ xl,
    const __bf16* __restrict__ yh, const __bf16* __restrict__ yl,
    const float* __restrict__ xn, const float* __restrict__ yn,
    float* __restrict__ out) {
  const int lane = threadIdx.x & 31;
  const int wave = threadIdx.x >> 5;
  const int half = lane >> 4;
  const int li   = lane & 15;

  const int waveM = wave & 3;
  const int waveN = wave >> 2;
  const int mBase = blockIdx.x * 128 + waveM * 32;
  const int nBase = blockIdx.y * 64 + waveN * 32;

  const v8f z = (v8f){0.f, 0.f, 0.f, 0.f, 0.f, 0.f, 0.f, 0.f};
  v8f acc00 = z, acc01 = z, acc10 = z, acc11 = z;

  const __bf16* a0h = xh + (size_t)(mBase + li) * RANK;
  const __bf16* a0l = xl + (size_t)(mBase + li) * RANK;
  const __bf16* a1h = xh + (size_t)(mBase + 16 + li) * RANK;
  const __bf16* a1l = xl + (size_t)(mBase + 16 + li) * RANK;
  const __bf16* b0h = yh + (size_t)(nBase + li) * RANK;
  const __bf16* b0l = yl + (size_t)(nBase + li) * RANK;
  const __bf16* b1h = yh + (size_t)(nBase + 16 + li) * RANK;
  const __bf16* b1l = yl + (size_t)(nBase + 16 + li) * RANK;

#pragma unroll
  for (int k = 0; k < RANK; k += 32) {
    v16bf A0h = load_fragA(a0h, k, half);
    v16bf A0l = load_fragA(a0l, k, half);
    v16bf A1h = load_fragA(a1h, k, half);
    v16bf A1l = load_fragA(a1l, k, half);
    v16bf B0h = load_fragB(b0h, k, half);
    v16bf B0l = load_fragB(b0l, k, half);
    v16bf B1h = load_fragB(b1h, k, half);
    v16bf B1l = load_fragB(b1l, k, half);

    acc00 = wmma_bf16(A0h, B0h, acc00);
    acc00 = wmma_bf16(A0h, B0l, acc00);
    acc00 = wmma_bf16(A0l, B0h, acc00);

    acc01 = wmma_bf16(A0h, B1h, acc01);
    acc01 = wmma_bf16(A0h, B1l, acc01);
    acc01 = wmma_bf16(A0l, B1h, acc01);

    acc10 = wmma_bf16(A1h, B0h, acc10);
    acc10 = wmma_bf16(A1h, B0l, acc10);
    acc10 = wmma_bf16(A1l, B0h, acc10);

    acc11 = wmma_bf16(A1h, B1h, acc11);
    acc11 = wmma_bf16(A1h, B1l, acc11);
    acc11 = wmma_bf16(A1l, B1h, acc11);
  }

  // Fused epilogue: d2 = xn[m] + yn[n] - 2*cross, clamped at 0.
  const float yn0 = yn[nBase + li];
  const float yn1 = yn[nBase + 16 + li];
#pragma unroll
  for (int i = 0; i < 8; ++i) {
    const int m0 = mBase + 8 * half + i;
    const int m1 = m0 + 16;
    const float xm0 = xn[m0];
    const float xm1 = xn[m1];
    const size_t r0 = (size_t)m0 * NROWS;
    const size_t r1 = (size_t)m1 * NROWS;
    out[r0 + nBase + li]      = fmaxf(xm0 + yn0 - 2.f * acc00[i], 0.f);
    out[r0 + nBase + 16 + li] = fmaxf(xm0 + yn1 - 2.f * acc01[i], 0.f);
    out[r1 + nBase + li]      = fmaxf(xm1 + yn0 - 2.f * acc10[i], 0.f);
    out[r1 + nBase + 16 + li] = fmaxf(xm1 + yn1 - 2.f * acc11[i], 0.f);
  }
}

extern "C" void kernel_launch(void* const* d_in, const int* in_sizes, int n_in,
                              void* d_out, int out_size, void* d_ws, size_t ws_size,
                              hipStream_t stream) {
  const float* x = (const float*)d_in[0];
  const float* y = (const float*)d_in[1];
  const float* L = (const float*)d_in[2];
  float* out = (float*)d_out;

  // Workspace layout: xh | xl | yh | yl (bf16, 2 MB each) | xn | yn (f32)
  const size_t projElems = (size_t)NROWS * RANK;
  __bf16* xh = (__bf16*)d_ws;
  __bf16* xl = xh + projElems;
  __bf16* yh = xl + projElems;
  __bf16* yl = yh + projElems;
  float* xn = (float*)(yl + projElems);
  float* yn = xn + NROWS;

  proj_split_kernel<<<dim3(NROWS / 128, 2), 256, 0, stream>>>(x, y, L, xh, xl, yh, yl);
  rownorm_kernel<<<dim3(NROWS / 256, 2), 256, 0, stream>>>(xh, xl, yh, yl, xn, yn);
  cross_kernel<<<dim3(NROWS / 128, NROWS / 64), 256, 0, stream>>>(xh, xl, yh, yl,
                                                                  xn, yn, out);
}
